// TransformerBlock_29437705846900
// MI455X (gfx1250) — matrix operand order repaired and checked by reading, not compiled, not run on hardware
//
#include <hip/hip_runtime.h>
#include <hip/hip_bf16.h>
#include <math.h>
#include <stdint.h>

// ---------------------------------------------------------------------------
// Transformer block for MI455X (gfx1250, wave32, WMMA).
// - All GEMMs: v_wmma_f32_16x16x32_f16 (f16 in, f32 accumulate)
// - LDS tiles transposed+padded so fragment reads are 2x ds_load_b128
// - V written transposed (vT[b,h,d,s]) by the QKV kernel
// - Attention K/V tiles staged to LDS with CDNA5 ASYNC copies
//   (global_load_async_to_lds_b128 + s_wait_asynccnt), shared by all 8 waves
// ---------------------------------------------------------------------------

typedef _Float16 f16;
typedef __attribute__((ext_vector_type(16))) _Float16 v16h;
typedef __attribute__((ext_vector_type(8)))  _Float16 v8h;
typedef __attribute__((ext_vector_type(8)))  float    v8f;

#define DIMC   1024
#define HEADS  16
#define DH     64
#define BB     4
#define SS     1024
#define NTOK   (BB * SS)     // 4096
#define FF     (4 * DIMC)    // 4096
#define TPAD   40            // padded row stride (halves), 80B rows: bank-clean
#define KPAD   72            // padded K-tile row stride (halves), 144B rows

// ----------------------------- CDNA5 helpers -------------------------------

static __device__ __forceinline__ v8f wmma16(v16h a, v16h b, v8f c) {
    // D = A(16x32 f16) * B(32x16 f16) + C(16x16 f32)
    return __builtin_amdgcn_wmma_f32_16x16x32_f16(
        /*neg_a=*/false, a, /*neg_b=*/false, b,
        /*c_mod=*/(short)0, c, /*reuse_a=*/false, /*reuse_b=*/false);
}

// Async global->LDS copy, 16B per lane, tracked by ASYNCcnt (ISA 15.18.3 #98).
static __device__ __forceinline__ void async_copy_b128(const void* gsrc, void* ldst) {
    unsigned           lds_off = (unsigned)(uintptr_t)ldst;           // addr[31:0]
    unsigned long long gaddr   = (unsigned long long)(uintptr_t)gsrc; // GV mode
    asm volatile("global_load_async_to_lds_b128 %0, %1, off"
                 :: "v"(lds_off), "v"(gaddr) : "memory");
}
static __device__ __forceinline__ void wait_asynccnt0() {
    asm volatile("s_wait_asynccnt 0" ::: "memory");
}

// A fragment (16x32, MxK) from a row-major f16 matrix; base -> (row0, k0).
// ISA layout: lo lanes K 0-7 / 16-23, hi lanes K 8-15 / 24-31 -> two B128s.
static __device__ __forceinline__ v16h load_a_frag(const f16* base, int stride, int lane) {
    int m  = lane & 15;
    int hi = lane >> 4;
    const f16* p = base + (size_t)m * stride + hi * 8;
    v16h a;
    *(v8h*)&a       = *(const v8h*)(p);       // halves 0..7
    *((v8h*)&a + 1) = *(const v8h*)(p + 16);  // halves 8..15
    return a;
}

// B fragment (32x16, KxN) where B(kk,n) = Msrc[n0+n][kk0+kk] (Msrc row-major,
// i.e. B = Msrc^T). Contiguous along kk -> two B128 loads. Works for global
// and for LDS tiles (stride = TPAD/KPAD).
static __device__ __forceinline__ v16h load_bT_frag(const f16* base, int stride,
                                                    int kk0, int lane) {
    int n  = lane & 15;
    int hi = lane >> 4;
    const f16* p = base + (size_t)n * stride + kk0 + hi * 8;
    v16h b;
    *(v8h*)&b       = *(const v8h*)(p);
    *((v8h*)&b + 1) = *(const v8h*)(p + 16);
    return b;
}

// ----------------------------- small kernels -------------------------------

__global__ __launch_bounds__(256) void cvt_f32_to_f16(const float* __restrict__ src,
                                                      f16* __restrict__ dst, int n) {
    for (int i = blockIdx.x * 256 + threadIdx.x; i < n; i += gridDim.x * 256)
        dst[i] = (f16)src[i];
}

__global__ __launch_bounds__(256) void layernorm_kernel(const float* __restrict__ x,
                                                        const float* __restrict__ w,
                                                        const float* __restrict__ b,
                                                        f16* __restrict__ out) {
    int row = blockIdx.x;
    const float* xr = x + (size_t)row * DIMC;
    float s = 0.f, s2 = 0.f;
    for (int i = threadIdx.x; i < DIMC; i += 256) {
        float v = xr[i];
        s += v; s2 += v * v;
    }
#pragma unroll
    for (int m = 16; m >= 1; m >>= 1) {
        s  += __shfl_xor(s,  m, 32);
        s2 += __shfl_xor(s2, m, 32);
    }
    __shared__ float rs[8], rs2[8];
    __shared__ float mu_s, rstd_s;
    int wave = threadIdx.x >> 5, lane = threadIdx.x & 31;
    if (lane == 0) { rs[wave] = s; rs2[wave] = s2; }
    __syncthreads();
    if (threadIdx.x == 0) {
        float S = 0.f, S2 = 0.f;
        for (int i = 0; i < 8; ++i) { S += rs[i]; S2 += rs2[i]; }
        float mu  = S / (float)DIMC;
        float var = S2 / (float)DIMC - mu * mu;
        mu_s = mu;
        rstd_s = rsqrtf(var + 1e-5f);
    }
    __syncthreads();
    float mu = mu_s, rstd = rstd_s;
    for (int i = threadIdx.x; i < DIMC; i += 256)
        out[(size_t)row * DIMC + i] = (f16)((xr[i] - mu) * rstd * w[i] + b[i]);
}

// ----------------------------- QKV projection ------------------------------
// Per head: q/k/v = h_head (NTOK x 64) @ W (64 x 64). Weights staged in LDS
// TRANSPOSED (WT[e][d], padded) so B fragments read as 2x ds_load_b128.
// q,k written row-major [tok][dim]; v written transposed vT[b,h,d,s].
// grid = (HEADS, NTOK/128), block = 256 (8 waves, each owns 16 rows).

__global__ __launch_bounds__(256) void qkv_kernel(const f16* __restrict__ h,
                                                  const f16* __restrict__ wq,
                                                  const f16* __restrict__ wk,
                                                  const f16* __restrict__ wv,
                                                  f16* __restrict__ q,
                                                  f16* __restrict__ k,
                                                  f16* __restrict__ vT) {
    __shared__ f16 ldsWT[3][64 * TPAD];   // 15 KB, transposed weight tiles
    int tid = threadIdx.x, lane = tid & 31, wave = tid >> 5;
    int head = blockIdx.x;
    int row0 = blockIdx.y * 128 + wave * 16;

    const f16* srcs[3] = { wq + head * 4096, wk + head * 4096, wv + head * 4096 };
    // transpose-stage: each chunk = one B128 global load + 8 b16 LDS stores
    for (int j = tid; j < 3 * 512; j += 256) {
        int w3 = j >> 9;
        int r  = j & 511;
        int d  = r >> 3, ch = r & 7;
        v8h g = *(const v8h*)(srcs[w3] + d * 64 + ch * 8);
#pragma unroll
        for (int e = 0; e < 8; ++e)
            ldsWT[w3][(ch * 8 + e) * TPAD + d] = g[e];
    }
    __syncthreads();

    const f16* hbase = h + (size_t)row0 * DIMC + head * 64;
    v16h a0 = load_a_frag(hbase,      DIMC, lane);
    v16h a1 = load_a_frag(hbase + 32, DIMC, lane);

    int n = lane & 15, hi = lane >> 4;
    int bb = row0 >> 10;          // row0 / S
    int s0 = row0 & (SS - 1);     // row0 % S

    // q and k: row-major outputs
    f16* outs[2] = { q, k };
#pragma unroll
    for (int w3 = 0; w3 < 2; ++w3) {
#pragma unroll
        for (int t = 0; t < 4; ++t) {
            v8f acc = {};
            v16h b0 = load_bT_frag(&ldsWT[w3][t * 16 * TPAD], TPAD, 0,  lane);
            v16h b1 = load_bT_frag(&ldsWT[w3][t * 16 * TPAD], TPAD, 32, lane);
            acc = wmma16(a0, b0, acc);
            acc = wmma16(a1, b1, acc);
            f16* op = outs[w3] + (size_t)row0 * DIMC + head * 64 + t * 16 + n;
#pragma unroll
            for (int r = 0; r < 8; ++r)
                op[(size_t)(r + hi * 8) * DIMC] = (f16)acc[r];
        }
    }
    // v: transposed output vT[((b*H+head)*DH + d)*S + s], contiguous in s
#pragma unroll
    for (int t = 0; t < 4; ++t) {
        v8f acc = {};
        v16h b0 = load_bT_frag(&ldsWT[2][t * 16 * TPAD], TPAD, 0,  lane);
        v16h b1 = load_bT_frag(&ldsWT[2][t * 16 * TPAD], TPAD, 32, lane);
        acc = wmma16(a0, b0, acc);
        acc = wmma16(a1, b1, acc);
        v8h pk;
#pragma unroll
        for (int r = 0; r < 8; ++r) pk[r] = (f16)acc[r];
        f16* vp = vT + ((size_t)((bb * HEADS + head) * DH + t * 16 + n)) * SS
                     + s0 + hi * 8;
        *(v8h*)vp = pk;   // one B128 store per lane per tile
    }
}

// ----------------------------- flash attention -----------------------------
// grid = (B*HEADS, S/128), block = 256 (8 waves x 16 query rows).
// Per 32-key block: K tile (32x64) and V tile (64x32, from vT) are staged to
// LDS ONCE PER BLOCK with async global->LDS B128 copies (ASYNCcnt), then all
// 8 waves read fragments from LDS. Online softmax; P re-laid out through LDS.

__global__ __launch_bounds__(256) void attn_kernel(const f16* __restrict__ q,
                                                   const f16* __restrict__ k,
                                                   const f16* __restrict__ vT,
                                                   f16* __restrict__ o) {
    __shared__ f16 ldsP[8][16 * TPAD];   // 10 KB per-wave P tiles (padded)
    __shared__ f16 ldsK[32 * KPAD];      // 4.5 KB shared K tile [tok][d]
    __shared__ f16 ldsV[64 * TPAD];      // 5 KB  shared V tile [d][s]
    int tid = threadIdx.x, lane = tid & 31, wave = tid >> 5;
    int pair = blockIdx.x;
    int b = pair >> 4, hd = pair & 15;
    int qrow = blockIdx.y * 128 + wave * 16;
    int n = lane & 15, hi = lane >> 4;

    const f16* Q = q + (size_t)(b * SS + qrow) * DIMC + hd * 64;
    v16h aq0 = load_a_frag(Q,      DIMC, lane);
    v16h aq1 = load_a_frag(Q + 32, DIMC, lane);

    const f16* Kh = k  + (size_t)(b * SS) * DIMC + hd * 64;           // [S][DIMC]
    const f16* Vh = vT + (size_t)(b * HEADS + hd) * DH * SS;          // [DH][S]

    // per-thread async staging assignments (16B chunks)
    int kr = tid >> 3, kch = tid & 7;   // K tile: 32 rows x 8 chunks
    int vr = tid >> 2, vch = tid & 3;   // V tile: 64 rows x 4 chunks

    v8f oacc[4] = {};
    float mrun[8], lsum[8];
#pragma unroll
    for (int r = 0; r < 8; ++r) { mrun[r] = -1e30f; lsum[r] = 0.f; }
    const float scale = 0.125f;   // 1/sqrt(64)

    f16* lp = ldsP[wave];

    for (int c0 = 0; c0 < SS; c0 += 32) {
        __syncthreads();   // previous iteration's LDS reads complete
        // async-stage K (32x64) and V (64x32) tiles: 2 B128 issues per thread
        async_copy_b128(Kh + (size_t)(c0 + kr) * DIMC + kch * 8,
                        &ldsK[kr * KPAD + kch * 8]);
        async_copy_b128(Vh + (size_t)vr * SS + c0 + vch * 8,
                        &ldsV[vr * TPAD + vch * 8]);
        wait_asynccnt0();  // own wave's async copies landed in LDS
        __syncthreads();   // all waves' copies landed

        // S = Q @ K^T  (two 16x16 logit tiles) from LDS K tile
        v8f s0 = {}, s1 = {};
        s0 = wmma16(aq0, load_bT_frag(ldsK,             KPAD, 0,  lane), s0);
        s0 = wmma16(aq1, load_bT_frag(ldsK,             KPAD, 32, lane), s0);
        s1 = wmma16(aq0, load_bT_frag(ldsK + 16 * KPAD, KPAD, 0,  lane), s1);
        s1 = wmma16(aq1, load_bT_frag(ldsK + 16 * KPAD, KPAD, 32, lane), s1);

        // online softmax per row (rows striped: M = r + 8*hi, group of 16 lanes)
#pragma unroll
        for (int r = 0; r < 8; ++r) {
            float x0 = s0[r] * scale, x1 = s1[r] * scale;
            float mx = fmaxf(x0, x1);
            mx = fmaxf(mx, __shfl_xor(mx, 1, 32));
            mx = fmaxf(mx, __shfl_xor(mx, 2, 32));
            mx = fmaxf(mx, __shfl_xor(mx, 4, 32));
            mx = fmaxf(mx, __shfl_xor(mx, 8, 32));
            float mN   = fmaxf(mrun[r], mx);
            float corr = __expf(mrun[r] - mN);
            mrun[r] = mN;
            float p0 = __expf(x0 - mN), p1 = __expf(x1 - mN);
            float ls = p0 + p1;
            ls += __shfl_xor(ls, 1, 32);
            ls += __shfl_xor(ls, 2, 32);
            ls += __shfl_xor(ls, 4, 32);
            ls += __shfl_xor(ls, 8, 32);
            lsum[r] = lsum[r] * corr + ls;
            lp[(r + hi * 8) * TPAD + n]      = (f16)p0;
            lp[(r + hi * 8) * TPAD + 16 + n] = (f16)p1;
            oacc[0][r] *= corr; oacc[1][r] *= corr;
            oacc[2][r] *= corr; oacc[3][r] *= corr;
        }

        // wave-local LDS RAW hazard: CDNA5 split DS counter wait
        asm volatile("s_wait_dscnt 0" ::: "memory");

        // P as A fragment: two B128 LDS reads (A halves are 2 contiguous runs)
        v16h ap;
        {
            int m = lane & 15;
            const f16* p = lp + m * TPAD + hi * 8;
            *(v8h*)&ap       = *(const v8h*)(p);
            *((v8h*)&ap + 1) = *(const v8h*)(p + 16);
        }
        // O += P @ V : V fragments from shared LDS tile
#pragma unroll
        for (int t = 0; t < 4; ++t) {
            v16h bv = load_bT_frag(ldsV + t * 16 * TPAD, TPAD, 0, lane);
            oacc[t] = wmma16(ap, bv, oacc[t]);
        }
    }

    f16* O = o + (size_t)(b * SS + qrow) * DIMC + hd * 64;
#pragma unroll
    for (int t = 0; t < 4; ++t)
#pragma unroll
        for (int r = 0; r < 8; ++r)
            O[(size_t)(r + hi * 8) * DIMC + t * 16 + n] = (f16)(oacc[t][r] / lsum[r]);
}

// ----------------------------- generic GEMM --------------------------------
// C[M,N] = A[M,K](f16) @ B[K,N](f16) + bias, epilogue: optional exact GELU,
// optional f32 residual add; writes f32 and/or f16.
// B tile staged TRANSPOSED in LDS (BT[n][kk], padded) -> fragment reads are
// 2x ds_load_b128, bank-conflict-free. Each wave owns a 32x64 tile.
// grid = (N/64, M/256), block = 256 (8 waves).

__global__ __launch_bounds__(256) void gemm_kernel(const f16* __restrict__ A,
                                                   const f16* __restrict__ Bw,
                                                   const float* __restrict__ bias,
                                                   const float* __restrict__ res,
                                                   float* __restrict__ outF32,
                                                   f16* __restrict__ outF16,
                                                   int M, int N, int K, int dogelu) {
    __shared__ f16 ldsBT[64 * TPAD];   // 5 KB transposed B tile
    int tid = threadIdx.x, lane = tid & 31, wave = tid >> 5;
    int col0 = blockIdx.x * 64;
    int row0 = blockIdx.y * 256 + wave * 32;
    int n = lane & 15, hi = lane >> 4;

    v8f acc[2][4] = {};
    int kk = tid >> 3, ch = tid & 7;
    const f16* A0 = A + (size_t)row0 * K;
    const f16* A1 = A + (size_t)(row0 + 16) * K;

    for (int k0 = 0; k0 < K; k0 += 32) {
        __syncthreads();
        // stage B tile transposed: one B128 load + 8 b16 scatter stores/thread
        v8h g = *(const v8h*)(Bw + (size_t)(k0 + kk) * N + col0 + ch * 8);
#pragma unroll
        for (int j = 0; j < 8; ++j)
            ldsBT[(ch * 8 + j) * TPAD + kk] = g[j];
        if (k0 + 32 < K)  // prefetch next B tile -> global_prefetch_b8
            __builtin_prefetch(Bw + (size_t)(k0 + 32 + kk) * N + col0 + ch * 8, 0, 1);
        __syncthreads();

        v16h a0 = load_a_frag(A0 + k0, K, lane);
        v16h a1 = load_a_frag(A1 + k0, K, lane);
#pragma unroll
        for (int t = 0; t < 4; ++t) {
            v16h bf = load_bT_frag(&ldsBT[t * 16 * TPAD], TPAD, 0, lane);
            acc[0][t] = wmma16(a0, bf, acc[0][t]);
            acc[1][t] = wmma16(a1, bf, acc[1][t]);
        }
    }

#pragma unroll
    for (int half = 0; half < 2; ++half) {
#pragma unroll
        for (int t = 0; t < 4; ++t) {
            int col = col0 + t * 16 + n;
            float bv = bias ? bias[col] : 0.f;
#pragma unroll
            for (int r = 0; r < 8; ++r) {
                int row = row0 + half * 16 + r + hi * 8;
                float val = acc[half][t][r] + bv;
                if (dogelu)
                    val = 0.5f * val * (1.f + erff(val * 0.70710678118654752f));
                if (res)    val += res[(size_t)row * N + col];
                if (outF32) outF32[(size_t)row * N + col] = val;
                if (outF16) outF16[(size_t)row * N + col] = (f16)val;
            }
        }
    }
}

// ----------------------------- host launcher -------------------------------

extern "C" void kernel_launch(void* const* d_in, const int* in_sizes, int n_in,
                              void* d_out, int out_size, void* d_ws, size_t ws_size,
                              hipStream_t stream) {
    (void)in_sizes; (void)n_in; (void)out_size; (void)ws_size;
    const float* x      = (const float*)d_in[0];
    const float* wq     = (const float*)d_in[1];
    const float* wk     = (const float*)d_in[2];
    const float* wv     = (const float*)d_in[3];
    const float* w_proj = (const float*)d_in[4];
    const float* b_proj = (const float*)d_in[5];
    const float* ln1_w  = (const float*)d_in[6];
    const float* ln1_b  = (const float*)d_in[7];
    const float* ln2_w  = (const float*)d_in[8];
    const float* ln2_b  = (const float*)d_in[9];
    const float* fc1_w  = (const float*)d_in[10];
    const float* fc1_b  = (const float*)d_in[11];
    const float* fc2_w  = (const float*)d_in[12];
    const float* fc2_b  = (const float*)d_in[13];

    char* ws = (char*)d_ws;
    size_t off = 0;
    auto take = [&](size_t bytes) -> char* {
        char* p = ws + off;
        off = (off + bytes + 255) & ~(size_t)255;
        return p;
    };
    f16*   h16    = (f16*)  take((size_t)NTOK * DIMC * 2);
    f16*   q16    = (f16*)  take((size_t)NTOK * DIMC * 2);
    f16*   k16    = (f16*)  take((size_t)NTOK * DIMC * 2);
    f16*   vT16   = (f16*)  take((size_t)NTOK * DIMC * 2);   // [B][H][DH][S]
    f16*   a16    = (f16*)  take((size_t)NTOK * DIMC * 2);
    float* x1     = (float*)take((size_t)NTOK * DIMC * 4);
    f16*   hff16  = (f16*)  take((size_t)NTOK * FF * 2);
    f16*   wq16   = (f16*)  take((size_t)HEADS * DH * DH * 2);
    f16*   wk16   = (f16*)  take((size_t)HEADS * DH * DH * 2);
    f16*   wv16   = (f16*)  take((size_t)HEADS * DH * DH * 2);
    f16*   wp16   = (f16*)  take((size_t)DIMC * DIMC * 2);
    f16*   wf1_16 = (f16*)  take((size_t)DIMC * FF * 2);
    f16*   wf2_16 = (f16*)  take((size_t)FF * DIMC * 2);

    // weight conversions f32 -> f16
    cvt_f32_to_f16<<<64,   256, 0, stream>>>(wq,     wq16,   HEADS * DH * DH);
    cvt_f32_to_f16<<<64,   256, 0, stream>>>(wk,     wk16,   HEADS * DH * DH);
    cvt_f32_to_f16<<<64,   256, 0, stream>>>(wv,     wv16,   HEADS * DH * DH);
    cvt_f32_to_f16<<<1024, 256, 0, stream>>>(w_proj, wp16,   DIMC * DIMC);
    cvt_f32_to_f16<<<2048, 256, 0, stream>>>(fc1_w,  wf1_16, DIMC * FF);
    cvt_f32_to_f16<<<2048, 256, 0, stream>>>(fc2_w,  wf2_16, FF * DIMC);

    // --- attention sub-block ---
    layernorm_kernel<<<NTOK, 256, 0, stream>>>(x, ln1_w, ln1_b, h16);
    qkv_kernel<<<dim3(HEADS, NTOK / 128), 256, 0, stream>>>(h16, wq16, wk16, wv16,
                                                            q16, k16, vT16);
    attn_kernel<<<dim3(BB * HEADS, SS / 128), 256, 0, stream>>>(q16, k16, vT16, a16);
    // x1 = x + attn @ w_proj + b_proj
    gemm_kernel<<<dim3(DIMC / 64, NTOK / 256), 256, 0, stream>>>(
        a16, wp16, b_proj, x, x1, nullptr, NTOK, DIMC, DIMC, 0);

    // --- FFN sub-block ---
    layernorm_kernel<<<NTOK, 256, 0, stream>>>(x1, ln2_w, ln2_b, h16);
    gemm_kernel<<<dim3(FF / 64, NTOK / 256), 256, 0, stream>>>(
        h16, wf1_16, fc1_b, nullptr, nullptr, hff16, NTOK, FF, DIMC, 1);
    gemm_kernel<<<dim3(DIMC / 64, NTOK / 256), 256, 0, stream>>>(
        hff16, wf2_16, fc2_b, x1, (float*)d_out, nullptr, NTOK, DIMC, FF, 0);
}
